// CompressedInteraction_3607772529172
// MI455X (gfx1250) — matrix operand order repaired and checked by reading, not compiled, not run on hardware
//
#include <hip/hip_runtime.h>

// ---------------------------------------------------------------------------
// CIN (xDeepFM compressed interaction) fused kernel for gfx1250 (MI455X).
// B=256, F=64, E=64, NFILTER=128, 3 layers.
// One workgroup per batch element; x0 / xk ping-pong kept in LDS (the huge
// (B,F*C,E) outer-product tensor is never materialized); W streamed from
// L2/HBM as bf16 (pre-converted), double-buffered into LDS with
// GLOBAL_LOAD_ASYNC_TO_LDS_B128 (software pipelined against the WMMAs),
// fed to V_WMMA_F32_16X16X32_BF16. A-operand built with V_PK_MUL_BF16.
// ---------------------------------------------------------------------------

typedef __attribute__((ext_vector_type(16))) __bf16 v16bf;
typedef __attribute__((ext_vector_type(8)))  float  v8f;
typedef __attribute__((ext_vector_type(4)))  int    int4v;

#define NB     256
#define FDIM   64
#define EDIM   64
#define CF     128      // NFILTER
#define NLAY   3

#define KCHUNK 128      // K rows of W staged per LDS chunk
#define WRS    144      // W-chunk LDS row stride in bf16 (128 + 16 pad; 288B, 16B-aligned rows)
#define XKRS   144      // xk LDS row stride in bf16 (128 + 16 pad)
#define XTRS   80       // x0 LDS row stride in bf16 (64 + 16 pad)

#if defined(__has_builtin)
#  if __has_builtin(__builtin_amdgcn_global_load_async_to_lds_b128) && \
      __has_builtin(__builtin_amdgcn_s_wait_asynccnt)
#    define USE_ASYNC_LDS 1
#  endif
#endif
#ifndef USE_ASYNC_LDS
#  define USE_ASYNC_LDS 0
#endif

__device__ __forceinline__ unsigned short f2bf(float f) {
  unsigned int x = __builtin_bit_cast(unsigned int, f);
  x += 0x7fffu + ((x >> 16) & 1u);   // round-to-nearest-even
  return (unsigned short)(x >> 16);
}
// packed bf16 multiply: one VOP3P instruction for 2 elements (RNE in HW)
__device__ __forceinline__ unsigned int pk_mul_bf16(unsigned int a, unsigned int b) {
  unsigned int d;
  asm("v_pk_mul_bf16 %0, %1, %2" : "=v"(d) : "v"(a), "v"(b));
  return d;
}

#if USE_ASYNC_LDS
// 16B global -> LDS async copy (GLOBAL_LOAD_ASYNC_TO_LDS_B128, ASYNCcnt)
__device__ __forceinline__ void async_copy16(const unsigned short* gsrc,
                                             unsigned short* ldst) {
  __builtin_amdgcn_global_load_async_to_lds_b128(
      (__attribute__((address_space(1))) int4v*)gsrc,
      (__attribute__((address_space(3))) int4v*)ldst, 0, 0);
}
#endif

// Stage W[o][k0..k0+KCHUNK) -> dst[o][0..KCHUNK): 2048 x 16B, coalesced.
// Each thread issues 8 (async) 16B copies.
__device__ __forceinline__ void stage_w(const unsigned short* __restrict__ Wl,
                                        int Kl, int k0,
                                        unsigned short* dst, int tid) {
#pragma unroll
  for (int i = 0; i < 8; ++i) {
    int idx = tid + i * 256;
    int o   = idx >> 4;
    int kq  = (idx & 15) * 8;           // bf16 offset within chunk
    const unsigned short* gsrc = Wl + (size_t)o * Kl + k0 + kq;
    unsigned short* ldst = dst + o * WRS + kq;
#if USE_ASYNC_LDS
    async_copy16(gsrc, ldst);
#else
    *(uint4*)ldst = *(const uint4*)gsrc;
#endif
  }
}

// ---------------------------------------------------------------------------
// W f32 -> bf16 conversion (runs every call; deterministic)
// ---------------------------------------------------------------------------
__global__ void f32_to_bf16_kernel(const float* __restrict__ in,
                                   unsigned short* __restrict__ out, int n) {
  int i = blockIdx.x * blockDim.x + threadIdx.x;
  if (i < n) out[i] = f2bf(in[i]);
}

// ---------------------------------------------------------------------------
// Fused 3-layer CIN. grid = B blocks, 256 threads (8 wave32).
// ---------------------------------------------------------------------------
__global__ __launch_bounds__(256) void cin_fused_kernel(
    const float* __restrict__ x,            // (B, F, E) f32
    const unsigned short* __restrict__ Wbf, // bf16: W0 | W1 | W2 (row-major (O,K))
    const float* __restrict__ Wa,           // (1, 3*CF) f32
    float* __restrict__ y)                  // (B,) f32
{
  extern __shared__ char smem_raw[];
  unsigned short* xT  = (unsigned short*)smem_raw;      // [EDIM][XTRS]  x0^T (e-major)
  unsigned short* xk0 = xT  + EDIM * XTRS;              // [EDIM][XKRS]
  unsigned short* xk1 = xk0 + EDIM * XKRS;              // [EDIM][XKRS]
  unsigned short* WcA = xk1 + EDIM * XKRS;              // [CF][WRS] W chunk buf A
  unsigned short* WcB = WcA + CF * WRS;                 // [CF][WRS] W chunk buf B
  float* pooled = (float*)(WcB + CF * WRS);             // [NLAY][CF]
  float* ysum   = pooled + NLAY * CF;                   // [1]

  const int b    = blockIdx.x;
  const int tid  = threadIdx.x;
  const int lane = tid & 31;
  const int wave = tid >> 5;
  const int wm   = wave & 3;    // m-tile: e rows [wm*16, wm*16+16)
  const int wn   = wave >> 2;   // n-half: o tiles [wn*4, wn*4+4)
  const int r    = lane & 15;   // row/col within tile
  const int h    = lane >> 4;   // lane half (K split per ISA A/B layouts)

  // init pooled sums
  for (int i = tid; i < NLAY * CF; i += 256) pooled[i] = 0.f;
  if (tid == 0) *ysum = 0.f;

  // load x[b] (F,E) f32 -> xT[e][f] bf16
  const float* xb = x + (size_t)b * FDIM * EDIM;
  for (int i = tid; i < FDIM * EDIM; i += 256) {
    int f = i >> 6, e = i & 63;
    xT[e * XTRS + f] = f2bf(xb[i]);
  }
  __syncthreads();

  const unsigned short* Wl = Wbf;
  for (int layer = 0; layer < NLAY; ++layer) {
    const int Cprev = (layer == 0) ? FDIM : CF;
    const int Kl    = FDIM * Cprev;
    const int nchunk = Kl / KCHUNK;
    const unsigned short* xkCur = (layer == 0) ? xT : ((layer == 1) ? xk0 : xk1);
    const int curRS = (layer == 0) ? XTRS : XKRS;
    unsigned short* xkNext = (layer & 1) ? xk1 : xk0;

    v8f acc[4];
#pragma unroll
    for (int i = 0; i < 4; ++i)
#pragma unroll
      for (int g = 0; g < 8; ++g) acc[i][g] = 0.f;

    // pipeline prologue: chunk 0 -> buf A
    stage_w(Wl, Kl, 0, WcA, tid);

    for (int c = 0; c < nchunk; ++c) {
      const int k0 = c * KCHUNK;
      unsigned short* buf = (c & 1) ? WcB : WcA;

      // issue next chunk into the other buffer, prefetch two chunks ahead
      if (c + 1 < nchunk) {
        stage_w(Wl, Kl, k0 + KCHUNK, (c & 1) ? WcA : WcB, tid);
        if (k0 + 2 * KCHUNK < Kl)
          __builtin_prefetch(Wl + (size_t)(tid & (CF - 1)) * Kl + k0 + 2 * KCHUNK, 0, 3);
      }
#if USE_ASYNC_LDS
      // async loads complete in order: <=8 pending means chunk c is resident
      if (c + 1 < nchunk) __builtin_amdgcn_s_wait_asynccnt(8);
      else                __builtin_amdgcn_s_wait_asynccnt(0);
#endif
      __syncthreads();   // chunk c visible to all waves

#pragma unroll
      for (int kb = 0; kb < KCHUNK; kb += 32) {
        const int kg = k0 + kb;
        const int f  = kg / Cprev;      // 32-block never straddles an f boundary
        const int c0 = kg % Cprev;

        // ---- Build A fragment (16x32 bf16) for m-tile wm ----
        // A[e][k] = x0[f][e] * xk[c0+k][e]; per-lane K layout per ISA:
        //   VGPR j<4 : k = h*8 + 2j + p ; VGPR j>=4 : k = 16 + h*8 + 2(j-4) + p
        const int e = wm * 16 + r;
        const unsigned int su = xT[e * XTRS + f];
        const unsigned int sd = su * 0x10001u;          // splat bf16 into both halves
        const unsigned short* xr = xkCur + e * curRS + c0 + h * 8;
        uint4 q0 = *(const uint4*)(xr);        // k = h*8 .. h*8+7
        uint4 q1 = *(const uint4*)(xr + 16);   // k = 16+h*8 .. 16+h*8+7
        unsigned int qq[8] = {q0.x, q0.y, q0.z, q0.w, q1.x, q1.y, q1.z, q1.w};
        union { unsigned int u[8]; v16bf v; } A;
#pragma unroll
        for (int j = 0; j < 8; ++j) A.u[j] = pk_mul_bf16(qq[j], sd);

        // ---- 4 n-tiles: load B fragment (32x16) from LDS, WMMA ----
#pragma unroll
        for (int nt = 0; nt < 4; ++nt) {
          const int o = (wn * 4 + nt) * 16 + r;
          // B layout: same K order as A: VGPR j<4 : K = h*8 + 2j + p,
          // VGPR j>=4 : K = 16 + h*8 + 2(j-4) + p; two 16B loads.
          const unsigned short* wr = buf + o * WRS + kb + h * 8;
          union { uint4 q[2]; v16bf v; } Bf;
          Bf.q[0] = *(const uint4*)(wr);
          Bf.q[1] = *(const uint4*)(wr + 16);
          acc[nt] = __builtin_amdgcn_wmma_f32_16x16x32_bf16(
              false, A.v, false, Bf.v, (short)0, acc[nt], false, false);
        }
      }
      __syncthreads();   // done reading buf before it is overwritten (c+2)
    }

    // ---- ReLU, write xkNext (bf16), accumulate pooled sums over e ----
#pragma unroll
    for (int nt = 0; nt < 4; ++nt) {
      const int o = (wn * 4 + nt) * 16 + r;   // C/D layout: N = r
      float psum = 0.f;
#pragma unroll
      for (int g = 0; g < 8; ++g) {           // C/D layout: M = g + 8*h
        float v = acc[nt][g];
        v = v > 0.f ? v : 0.f;
        psum += v;
        const int e = wm * 16 + g + h * 8;
        xkNext[e * XKRS + o] = f2bf(v);
      }
      atomicAdd(&pooled[layer * CF + o], psum);
    }
    __syncthreads();

    Wl += (size_t)CF * Kl;
  }

  // ---- y[b] = pooled . Wa ----
  if (tid < CF) {
    float p = pooled[tid]            * Wa[tid]
            + pooled[CF + tid]       * Wa[CF + tid]
            + pooled[2 * CF + tid]   * Wa[2 * CF + tid];
    atomicAdd(ysum, p);
  }
  __syncthreads();
  if (tid == 0) y[b] = *ysum;
}

// ---------------------------------------------------------------------------
extern "C" void kernel_launch(void* const* d_in, const int* in_sizes, int n_in,
                              void* d_out, int out_size, void* d_ws, size_t ws_size,
                              hipStream_t stream) {
  (void)in_sizes; (void)n_in; (void)out_size; (void)ws_size;
  const float* x  = (const float*)d_in[0];
  const float* W0 = (const float*)d_in[1];
  const float* W1 = (const float*)d_in[2];
  const float* W2 = (const float*)d_in[3];
  const float* Wa = (const float*)d_in[4];
  float* y = (float*)d_out;

  unsigned short* wbf = (unsigned short*)d_ws;
  const int n0 = CF * FDIM * FDIM;  // 128*4096
  const int n1 = CF * FDIM * CF;    // 128*8192
  const int n2 = n1;

  f32_to_bf16_kernel<<<(n0 + 255) / 256, 256, 0, stream>>>(W0, wbf, n0);
  f32_to_bf16_kernel<<<(n1 + 255) / 256, 256, 0, stream>>>(W1, wbf + n0, n1);
  f32_to_bf16_kernel<<<(n2 + 255) / 256, 256, 0, stream>>>(W2, wbf + n0 + n1, n2);

  const size_t shbytes =
      (size_t)(EDIM * XTRS + 2 * EDIM * XKRS + 2 * CF * WRS) * sizeof(unsigned short)
      + (size_t)(NLAY * CF + 1) * sizeof(float);

  cin_fused_kernel<<<NB, 256, shbytes, stream>>>(x, wbf, Wa, y);
}
